// SelfAttentionLayer_74715251081388
// MI455X (gfx1250) — matrix operand order repaired and checked by reading, not compiled, not run on hardware
//
#include <hip/hip_runtime.h>

// ---- types ----
typedef __attribute__((ext_vector_type(16))) _Float16   v16h;
typedef __attribute__((ext_vector_type(8)))  float      v8f;
typedef __attribute__((ext_vector_type(4)))  float      f32x4;
typedef __attribute__((ext_vector_type(4)))  unsigned   u32x4;

#define B_TOT   32768
#define F_IN    512
#define D_HEAD  64
#define KB_CNT  16              // 512 / 32
#define LDS_STRIDE 68           // 64 + 4 pad (64 banks; 68%64=4 -> conflict-free)
#define PER_WAVE_F (48*LDS_STRIDE + 64)   // K,Q,V tiles (3*16 rows) + 64 attn slots

// ---------------------------------------------------------------------------
// Prep: swizzle Wk/Wq/Wv (fp32 [512][64]) into f16 WMMA B-fragments in d_ws.
// B operand of v_wmma_f32_16x16x32_f16 is 32x16 (KxN), 8 VGPRs/lane (v16h):
//   VGPR j, lanes 0-15 : K = 2j, 2j+1 (j<4) / 2j+8, 2j+9 (j>=4) (packed lo/hi)
//   VGPR j, lanes16-31 : K = 8+2j, 8+2j+1 (j<4) / 16+2j, 16+2j+1 (j>=4)
// Fragment (w, c, kb) stored lane-major: 32 lanes x 8 dwords (32B/lane) so the
// GEMM kernel loads each fragment with two fully-coalesced global_load_b128.
// Total ws use: 3*4*16 frags * 1KB = 192 KB.
// ---------------------------------------------------------------------------
__global__ void prep_weights(const float* __restrict__ Wk,
                             const float* __restrict__ Wq,
                             const float* __restrict__ Wv,
                             unsigned* __restrict__ frag)
{
    int t = blockIdx.x * blockDim.x + threadIdx.x;     // 49152 threads total
    if (t >= 3*4*16*32*8) return;
    int j    =  t        & 7;
    int lane = (t >> 3)  & 31;
    int kb   = (t >> 8)  & 15;
    int c    = (t >> 12) & 3;
    int w    =  t >> 14;
    const float* W = (w == 0) ? Wk : ((w == 1) ? Wq : Wv);

    int k0  = kb*32 + ((lane < 16) ? 0 : 8) + 2*j + ((j < 4) ? 0 : 8);
    int col = c*16 + (lane & 15);
    _Float16 h0 = (_Float16)W[(size_t)k0      * D_HEAD + col];
    _Float16 h1 = (_Float16)W[(size_t)(k0+1)  * D_HEAD + col];
    unsigned p = ((unsigned)__builtin_bit_cast(unsigned short, h1) << 16)
               |  (unsigned)__builtin_bit_cast(unsigned short, h0);
    int fragId = (w*4 + c)*16 + kb;
    frag[(size_t)fragId*256 + lane*8 + j] = p;
}

// ---------------------------------------------------------------------------
// Fused projections + attention + softmax + maxpool.
// One wave = TWO adjacent 16-row M-tiles (8 batches): each shared-weight B
// fragment load feeds two WMMAs, halving L2 traffic for the shared operand
// (2048 waves -> 1024 waves * 192KB = 192MB through L2/WGP$, under the
// 64MB/23.3TB/s HBM floor). 24 f32 accumulators (192 VGPRs) + 2 A frags.
//
// A fragment (16x32 f16) built per k-step from fp32 drug:
//   lanes 0-15  (M = lane)    : v0..v3 <- K 0..7,  v4..v7 <- K 16..23
//   lanes 16-31 (M = lane-16) : v0..v3 <- K 8..15, v4..v7 <- K 24..31
// Each 128B drug line is fully consumed within one k-step (no reuse), so NT
// loads keep the 64MB stream from evicting the resident weight fragments.
// ---------------------------------------------------------------------------
__global__ void __launch_bounds__(128)
attn_kernel(const float* __restrict__ drug,
            const unsigned* __restrict__ frag,
            float* __restrict__ out)
{
    __shared__ float smem[4 * PER_WAVE_F];   // 53,248 B

    const int lane  = threadIdx.x & 31;
    const int wave  = threadIdx.x >> 5;
    const int tile0 = (blockIdx.x * 4 + wave) * 2;     // this wave: tile0, tile0+1
    const int runsel = (lane < 16) ? 0 : 8;
    const float* arow0 = drug + (size_t)(tile0*16 + (lane & 15)) * F_IN;
    const float* arow1 = arow0 + (size_t)16 * F_IN;

    v8f acc[24] = {};

    for (int kb = 0; kb < KB_CNT; ++kb) {
        const float* p0 = arow0 + kb*32 + runsel;
        const float* p1 = arow1 + kb*32 + runsel;
        f32x4 r0 = __builtin_nontemporal_load((const f32x4*)(p0));
        f32x4 r1 = __builtin_nontemporal_load((const f32x4*)(p0 + 4));
        f32x4 r2 = __builtin_nontemporal_load((const f32x4*)(p0 + 16));
        f32x4 r3 = __builtin_nontemporal_load((const f32x4*)(p0 + 20));
        f32x4 s0 = __builtin_nontemporal_load((const f32x4*)(p1));
        f32x4 s1 = __builtin_nontemporal_load((const f32x4*)(p1 + 4));
        f32x4 s2 = __builtin_nontemporal_load((const f32x4*)(p1 + 16));
        f32x4 s3 = __builtin_nontemporal_load((const f32x4*)(p1 + 20));

        v16h a0, a1;
        a0[0]=(_Float16)r0.x;  a0[1]=(_Float16)r0.y;  a0[2]=(_Float16)r0.z;  a0[3]=(_Float16)r0.w;
        a0[4]=(_Float16)r1.x;  a0[5]=(_Float16)r1.y;  a0[6]=(_Float16)r1.z;  a0[7]=(_Float16)r1.w;
        a0[8]=(_Float16)r2.x;  a0[9]=(_Float16)r2.y;  a0[10]=(_Float16)r2.z; a0[11]=(_Float16)r2.w;
        a0[12]=(_Float16)r3.x; a0[13]=(_Float16)r3.y; a0[14]=(_Float16)r3.z; a0[15]=(_Float16)r3.w;
        a1[0]=(_Float16)s0.x;  a1[1]=(_Float16)s0.y;  a1[2]=(_Float16)s0.z;  a1[3]=(_Float16)s0.w;
        a1[4]=(_Float16)s1.x;  a1[5]=(_Float16)s1.y;  a1[6]=(_Float16)s1.z;  a1[7]=(_Float16)s1.w;
        a1[8]=(_Float16)s2.x;  a1[9]=(_Float16)s2.y;  a1[10]=(_Float16)s2.z; a1[11]=(_Float16)s2.w;
        a1[12]=(_Float16)s3.x; a1[13]=(_Float16)s3.y; a1[14]=(_Float16)s3.z; a1[15]=(_Float16)s3.w;

        #pragma unroll
        for (int f = 0; f < 12; ++f) {
            const u32x4* bp = (const u32x4*)(frag + (size_t)(f*16 + kb)*256 + lane*8);
            union { u32x4 u[2]; v16h h; } bb;
            bb.u[0] = bp[0];
            bb.u[1] = bp[1];
            acc[f]      = __builtin_amdgcn_wmma_f32_16x16x32_f16(
                              false, a0, false, bb.h, (short)0, acc[f],      false, false);
            acc[12 + f] = __builtin_amdgcn_wmma_f32_16x16x32_f16(
                              false, a1, false, bb.h, (short)0, acc[12 + f], false, false);
        }
    }

    // ---- epilogue: two tiles sequentially, reusing this wave's LDS region ----
    float* base   = smem + wave * PER_WAVE_F;
    const int rsel = (lane < 16) ? 0 : 8;
    const int coll = lane & 15;
    float* Kb   = base;
    float* Qb   = base + 16 * LDS_STRIDE;
    float* Vb   = base + 32 * LDS_STRIDE;
    float* attn = base + 48 * LDS_STRIDE;   // 64 floats: [b][n][m]

    #pragma unroll
    for (int t = 0; t < 2; ++t) {
        const int tile = tile0 + t;

        // spill K,Q,V tiles (C/D layout: VGPR r -> row r / 8+r, col = c*16+lane%16)
        #pragma unroll
        for (int f = 0; f < 12; ++f) {
            int w = f >> 2, c = f & 3;
            float* mat = base + w * 16 * LDS_STRIDE;
            #pragma unroll
            for (int r = 0; r < 8; ++r)
                mat[(r + rsel) * LDS_STRIDE + c*16 + coll] = acc[t*12 + f][r];
        }

        // scores + softmax: 16 lanes, one (batch,n) row each (no 1/sqrt(d), per ref)
        if (lane < 16) {
            int b = lane >> 2, n = lane & 3;
            const float* kr = Kb + (b*4 + n) * LDS_STRIDE;
            float sc[4];
            #pragma unroll
            for (int m = 0; m < 4; ++m) {
                const float* qr = Qb + (b*4 + m) * LDS_STRIDE;
                float acc_s = 0.f;
                for (int d = 0; d < D_HEAD; ++d) acc_s += kr[d] * qr[d];
                sc[m] = acc_s;
            }
            float mx = fmaxf(fmaxf(sc[0], sc[1]), fmaxf(sc[2], sc[3]));
            float e0 = __expf(sc[0]-mx), e1 = __expf(sc[1]-mx),
                  e2 = __expf(sc[2]-mx), e3 = __expf(sc[3]-mx);
            float inv = 1.f / (e0 + e1 + e2 + e3);
            attn[b*16 + n*4 + 0] = e0 * inv;
            attn[b*16 + n*4 + 1] = e1 * inv;
            attn[b*16 + n*4 + 2] = e2 * inv;
            attn[b*16 + n*4 + 3] = e3 * inv;
        }
        // same-wave DS ops are in-order: reads below see writes above, and the
        // next t-iteration's spill cannot pass these reads (WAR kept in-order).

        // out = max_n (attn @ V), coalesced stores (lane = d)
        #pragma unroll
        for (int b = 0; b < 4; ++b) {
            size_t obase = (size_t)(tile*4 + b) * D_HEAD;
            #pragma unroll
            for (int part = 0; part < 2; ++part) {
                int d = lane + 32*part;
                float best = -3.402823466e38f;
                #pragma unroll
                for (int n = 0; n < 4; ++n) {
                    float sv = 0.f;
                    #pragma unroll
                    for (int m = 0; m < 4; ++m)
                        sv += attn[b*16 + n*4 + m] * Vb[(b*4 + m)*LDS_STRIDE + d];
                    best = fmaxf(best, sv);
                }
                out[obase + d] = best;
            }
        }
    }
}

extern "C" void kernel_launch(void* const* d_in, const int* in_sizes, int n_in,
                              void* d_out, int out_size, void* d_ws, size_t ws_size,
                              hipStream_t stream)
{
    const float* drug = (const float*)d_in[0];
    const float* Wk   = (const float*)d_in[1];
    const float* Wq   = (const float*)d_in[2];
    const float* Wv   = (const float*)d_in[3];
    unsigned* frag = (unsigned*)d_ws;          // needs 192 KB of scratch

    prep_weights<<<(3*4*16*32*8 + 255)/256, 256, 0, stream>>>(Wk, Wq, Wv, frag);
    attn_kernel<<<B_TOT/16/8, 128, 0, stream>>>(drug, frag, (float*)d_out);  // 256 blocks
}
